// TransformerBlock_16149077033205
// MI455X (gfx1250) — matrix operand order repaired and checked by reading, not compiled, not run on hardware
//
#include <hip/hip_runtime.h>
#include <hip/hip_bf16.h>
#include <cstdint>

typedef _Float16 half_t;
typedef __attribute__((ext_vector_type(16))) _Float16 v16h;
typedef __attribute__((ext_vector_type(8)))  _Float16 v8h;
typedef __attribute__((ext_vector_type(8)))  float    v8f;

// ---------------------------------------------------------------------------
// Conversion / transpose kernels (one-time weight prep, f32 -> f16)
// ---------------------------------------------------------------------------
__global__ __launch_bounds__(256) void cvt_f16_kernel(const float* __restrict__ in,
                                                      half_t* __restrict__ out, int n) {
  int i = blockIdx.x * 256 + threadIdx.x;
  if (i < n) out[i] = (half_t)in[i];
}

// Bt[n*K + k] = B[k*N + n]
__global__ __launch_bounds__(256) void transpose_cvt_kernel(const float* __restrict__ B,
                                                            half_t* __restrict__ Bt,
                                                            int K, int N) {
  int k = blockIdx.x * 16 + (threadIdx.x & 15);
  int n = blockIdx.y * 16 + (threadIdx.x >> 4);
  if (k < K && n < N) Bt[(size_t)n * K + k] = (half_t)B[(size_t)k * N + n];
}

// Bt[n*K + k] = Q[k*N + n] - Kw[k*N + n]   (fuses q-k projection difference)
__global__ __launch_bounds__(256) void transpose_sub_cvt_kernel(const float* __restrict__ Q,
                                                                const float* __restrict__ Kw,
                                                                half_t* __restrict__ Bt,
                                                                int K, int N) {
  int k = blockIdx.x * 16 + (threadIdx.x & 15);
  int n = blockIdx.y * 16 + (threadIdx.x >> 4);
  if (k < K && n < N)
    Bt[(size_t)n * K + k] = (half_t)(Q[(size_t)k * N + n] - Kw[(size_t)k * N + n]);
}

// ---------------------------------------------------------------------------
// WMMA GEMM: D[M,N] = A[M,K](f16,row-major) * Bt[N,K](f16, pre-transposed B)
// MODE 0: f16 out (+ optional bias)
// MODE 1: f16 out, bias + ReLU
// MODE 2: f32 out, bias + residual (final layer)
//
// Register-blocked: one wave computes a 32x64 tile = 2 (M) x 4 (N) WMMA tiles,
// 8 independent v_wmma_f32_16x16x32_f16 per K-chunk with shared A fragments
// (~21 FLOP/byte of fragment traffic). Block = 8 waves stacked in M ->
// block tile 256x64. Grid: (N/64, M/256).
// ---------------------------------------------------------------------------
template <int MODE>
__global__ __launch_bounds__(256)
void gemm_wmma_kernel(const half_t* __restrict__ A, const half_t* __restrict__ Bt,
                      const float* __restrict__ bias, const float* __restrict__ resid,
                      half_t* __restrict__ Oh, float* __restrict__ Of,
                      int M, int K, int N) {
  const int wave = threadIdx.x >> 5;
  const int lane = threadIdx.x & 31;
  const int col  = lane & 15;   // A-row / B-col / D-col index for this lane
  const int hi   = lane >> 4;   // half-wave group
  const int m0   = (blockIdx.y * 8 + wave) * 32;
  const int n0   = blockIdx.x * 64;
  if (m0 >= M) return;  // wave-uniform: EXEC stays all-ones for live waves

  const half_t* __restrict__ arow[2];
  const half_t* __restrict__ brow[4];
#pragma unroll
  for (int i = 0; i < 2; ++i) arow[i] = A  + (size_t)(m0 + i * 16 + col) * K;
#pragma unroll
  for (int j = 0; j < 4; ++j) brow[j] = Bt + (size_t)(n0 + j * 16 + col) * K;

  v8f acc[2][4] = {};
  for (int k0 = 0; k0 < K; k0 += 32) {
    if (k0 + 32 < K) {
#pragma unroll
      for (int i = 0; i < 2; ++i) __builtin_prefetch(arow[i] + k0 + 32, 0, 1);
#pragma unroll
      for (int j = 0; j < 4; ++j) __builtin_prefetch(brow[j] + k0 + 32, 0, 1);
    }
    v16h a[2], b[4];
    // A 16x32 f16 fragment: half-wave hi holds K = hi*8..+7 then 16+hi*8..+7
#pragma unroll
    for (int i = 0; i < 2; ++i) {
      v8h lo = *(const v8h*)(arow[i] + k0 + hi * 8);
      v8h hh = *(const v8h*)(arow[i] + k0 + 16 + hi * 8);
#pragma unroll
      for (int e = 0; e < 8; ++e) { a[i][e] = lo[e]; a[i][8 + e] = hh[e]; }
    }
    // B 32x16 f16 fragment: half-wave hi holds K = hi*8..+7 then 16+hi*8..+7
#pragma unroll
    for (int j = 0; j < 4; ++j) {
      v8h lo = *(const v8h*)(brow[j] + k0 + hi * 8);
      v8h hh = *(const v8h*)(brow[j] + k0 + 16 + hi * 8);
#pragma unroll
      for (int e = 0; e < 8; ++e) { b[j][e] = lo[e]; b[j][8 + e] = hh[e]; }
    }
#pragma unroll
    for (int i = 0; i < 2; ++i)
#pragma unroll
      for (int j = 0; j < 4; ++j)
        acc[i][j] = __builtin_amdgcn_wmma_f32_16x16x32_f16(
            /*neg_a=*/false, a[i], /*neg_b=*/false, b[j],
            /*c_mod=*/(short)0, acc[i][j], /*reuse_a=*/false, /*reuse_b=*/false);
  }

#pragma unroll
  for (int j = 0; j < 4; ++j) {
    const int cidx = n0 + j * 16 + col;
    const float bb = bias ? bias[cidx] : 0.0f;
#pragma unroll
    for (int i = 0; i < 2; ++i) {
#pragma unroll
      for (int r = 0; r < 8; ++r) {
        const int row = m0 + i * 16 + hi * 8 + r;  // C/D layout: VGPR r -> M = r + hi*8
        float v = acc[i][j][r] + bb;
        if (MODE == 1) v = v > 0.0f ? v : 0.0f;
        const size_t idx = (size_t)row * N + cidx;
        if (MODE == 2) {
          Of[idx] = v + resid[idx];
        } else {
          Oh[idx] = (half_t)v;
        }
      }
    }
  }
}

// ---------------------------------------------------------------------------
// Per-column softmax statistics over axis 0: max and sum(exp(x*scale - max))
// One block per column; column stride = N.
// ---------------------------------------------------------------------------
__global__ __launch_bounds__(256)
void colstats_kernel(const half_t* __restrict__ Hm, int rows, int N, float scale,
                     float* __restrict__ cmax, float* __restrict__ csum) {
  const int c = blockIdx.x;
  __shared__ float red[256];
  float m = -1e30f;
  for (int r = threadIdx.x; r < rows; r += 256)
    m = fmaxf(m, (float)Hm[(size_t)r * N + c] * scale);
  red[threadIdx.x] = m;
  __syncthreads();
  for (int s = 128; s > 0; s >>= 1) {
    if (threadIdx.x < s) red[threadIdx.x] = fmaxf(red[threadIdx.x], red[threadIdx.x + s]);
    __syncthreads();
  }
  m = red[0];
  __syncthreads();
  float sum = 0.0f;
  for (int r = threadIdx.x; r < rows; r += 256)
    sum += __expf((float)Hm[(size_t)r * N + c] * scale - m);
  red[threadIdx.x] = sum;
  __syncthreads();
  for (int s = 128; s > 0; s >>= 1) {
    if (threadIdx.x < s) red[threadIdx.x] += red[threadIdx.x + s];
    __syncthreads();
  }
  if (threadIdx.x == 0) { cmax[c] = m; csum[c] = red[0]; }
}

// ---------------------------------------------------------------------------
// Fused: attn1 (group softmax) + attn2 (tiled softmax) , PE , res = attn*(v+pe0)
// res written f16 for the final WMMA GEMM.
// ---------------------------------------------------------------------------
__global__ __launch_bounds__(256)
void fuse_res_kernel(const half_t* __restrict__ h1, const half_t* __restrict__ h2,
                     const half_t* __restrict__ v16,
                     const float* __restrict__ m1, const float* __restrict__ s1,
                     const float* __restrict__ m2, const float* __restrict__ s2,
                     const float* __restrict__ features, const float* __restrict__ lpe,
                     half_t* __restrict__ res, int Nrows, int H, int C) {
  const size_t i = (size_t)blockIdx.x * 256 + threadIdx.x;
  if (i >= (size_t)Nrows * H) return;
  const int r = (int)(i / H);
  const int c = (int)(i - (size_t)r * H);
  const float S1 = 0.08838834764831845f;  // 1/sqrt(128)
  const float S2 = 0.03125f;              // 1/sqrt(1024)
  const float a1 = __expf((float)h1[i] * S1 - m1[c]) / s1[c];
  const float a2 = __expf((float)h2[(size_t)(r & 2047) * H + c] * S2 - m2[c]) / s2[c];
  // PE: div = 100*f0 / 1000^(cf/512); emb = sin/cos; pe = (emb+lpe)*lpe
  const float f0 = features[(size_t)r * C];
  const int  cf   = c & 511;
  const float div = 100.0f * f0 * __expf(-(float)cf * (6.907755278982137f / 512.0f));
  const float emb = (c >= 512) ? __cosf(div) : __sinf(div);
  const float L  = lpe[0];
  const float pe = (emb + L) * L;
  res[i] = (half_t)((a1 + a2) * ((float)v16[i] + pe));
}

// ---------------------------------------------------------------------------
// Launcher
// ---------------------------------------------------------------------------
extern "C" void kernel_launch(void* const* d_in, const int* in_sizes, int n_in,
                              void* d_out, int out_size, void* d_ws, size_t ws_size,
                              hipStream_t stream) {
  (void)in_sizes; (void)n_in; (void)out_size; (void)ws_size;
  const int NN = 16384, CC = 64, HH = 1024, GS = 128, M2 = 2048;
  const int GM = NN * 8;                 // group-matmul rows (N*NUM, cols=128)
  const int GF2_OFF = (8 - 1) * (NN / 8); // row 14336: gf2 = x[14336:, :]

  const float* features = (const float*)d_in[0];
  const float* fc1_w = (const float*)d_in[1];  const float* fc1_b = (const float*)d_in[2];
  const float* fc2_w = (const float*)d_in[3];  const float* fc2_b = (const float*)d_in[4];
  const float* g1_w  = (const float*)d_in[5];  const float* g1_b  = (const float*)d_in[6];
  const float* g2_w  = (const float*)d_in[7];  const float* g2_b  = (const float*)d_in[8];
  const float* gg1_w = (const float*)d_in[9];  const float* gg1_b = (const float*)d_in[10];
  const float* gg2_w = (const float*)d_in[11]; const float* gg2_b = (const float*)d_in[12];
  const float* wq  = (const float*)d_in[13];
  const float* wk  = (const float*)d_in[14];
  const float* wv  = (const float*)d_in[15];
  const float* wq2 = (const float*)d_in[16];
  const float* wk2 = (const float*)d_in[17];
  const float* lpe = (const float*)d_in[18];
  float* out = (float*)d_out;

  // ---- workspace carve-up (256B aligned) ----
  uint8_t* ws = (uint8_t*)d_ws;
  size_t cur = 0;
  auto alloc = [&](size_t bytes) {
    void* p = ws + cur;
    cur += (bytes + 255) & ~(size_t)255;
    return p;
  };
  half_t* x16    = (half_t*)alloc((size_t)NN * HH * 2);
  half_t* t1     = (half_t*)alloc((size_t)NN * HH * 2);
  half_t* u1     = (half_t*)alloc((size_t)NN * HH * 2);
  half_t* h1     = (half_t*)alloc((size_t)NN * HH * 2);
  half_t* t2     = (half_t*)alloc((size_t)M2 * HH * 2);
  half_t* u2     = (half_t*)alloc((size_t)M2 * HH * 2);
  half_t* h2     = (half_t*)alloc((size_t)M2 * HH * 2);
  half_t* v16    = (half_t*)alloc((size_t)NN * HH * 2);
  half_t* res16  = (half_t*)alloc((size_t)NN * HH * 2);
  half_t* feat16 = (half_t*)alloc((size_t)NN * CC * 2);
  half_t* fc1wT  = (half_t*)alloc((size_t)CC * HH * 2);
  half_t* dW2T   = (half_t*)alloc((size_t)GS * GS * 2);
  half_t* gg1T   = (half_t*)alloc((size_t)GS * GS * 2);
  half_t* gg2T   = (half_t*)alloc((size_t)GS * GS * 2);
  half_t* dWT    = (half_t*)alloc((size_t)HH * HH * 2);
  half_t* g1T    = (half_t*)alloc((size_t)HH * HH * 2);
  half_t* g2T    = (half_t*)alloc((size_t)HH * HH * 2);
  half_t* wvT    = (half_t*)alloc((size_t)HH * HH * 2);
  half_t* fc2wT  = (half_t*)alloc((size_t)HH * CC * 2);
  float*  cmax1  = (float*)alloc((size_t)HH * 4);
  float*  csum1  = (float*)alloc((size_t)HH * 4);
  float*  cmax2  = (float*)alloc((size_t)HH * 4);
  float*  csum2  = (float*)alloc((size_t)HH * 4);

  // ---- weight prep ----
  cvt_f16_kernel<<<(NN * CC + 255) / 256, 256, 0, stream>>>(features, feat16, NN * CC);
  transpose_cvt_kernel<<<dim3(CC / 16, HH / 16), 256, 0, stream>>>(fc1_w, fc1wT, CC, HH);
  transpose_sub_cvt_kernel<<<dim3(GS / 16, GS / 16), 256, 0, stream>>>(wq2, wk2, dW2T, GS, GS);
  transpose_cvt_kernel<<<dim3(GS / 16, GS / 16), 256, 0, stream>>>(gg1_w, gg1T, GS, GS);
  transpose_cvt_kernel<<<dim3(GS / 16, GS / 16), 256, 0, stream>>>(gg2_w, gg2T, GS, GS);
  transpose_sub_cvt_kernel<<<dim3(HH / 16, HH / 16), 256, 0, stream>>>(wq, wk, dWT, HH, HH);
  transpose_cvt_kernel<<<dim3(HH / 16, HH / 16), 256, 0, stream>>>(g1_w, g1T, HH, HH);
  transpose_cvt_kernel<<<dim3(HH / 16, HH / 16), 256, 0, stream>>>(g2_w, g2T, HH, HH);
  transpose_cvt_kernel<<<dim3(HH / 16, HH / 16), 256, 0, stream>>>(wv, wvT, HH, HH);
  transpose_cvt_kernel<<<dim3(HH / 16, HH / 16), 256, 0, stream>>>(fc2_w, fc2wT, HH, CC);

  // ---- stage 1: x = features @ fc1_w + fc1_b  (16384x64 @ 64x1024) ----
  gemm_wmma_kernel<0><<<dim3(HH / 64, NN / 256), 256, 0, stream>>>(
      feat16, fc1wT, fc1_b, nullptr, x16, nullptr, NN, CC, HH);

  // ---- group branch: t1 = xg @ (wq2-wk2); u1 = relu(t1@gg1+b); h1 = u1@gg2+b ----
  // (131072 x 128) with 128x128 weights (same memory as (16384 x 1024))
  gemm_wmma_kernel<0><<<dim3(GS / 64, GM / 256), 256, 0, stream>>>(
      x16, dW2T, nullptr, nullptr, t1, nullptr, GM, GS, GS);
  gemm_wmma_kernel<1><<<dim3(GS / 64, GM / 256), 256, 0, stream>>>(
      t1, gg1T, gg1_b, nullptr, u1, nullptr, GM, GS, GS);
  gemm_wmma_kernel<0><<<dim3(GS / 64, GM / 256), 256, 0, stream>>>(
      u1, gg2T, gg2_b, nullptr, h1, nullptr, GM, GS, GS);

  // ---- tail branch: gf2 = x[14336:, :]; t2 = gf2@(wq-wk); u2 = relu(t2@g1+b); h2 = u2@g2+b ----
  const half_t* gf2 = x16 + (size_t)GF2_OFF * HH;
  gemm_wmma_kernel<0><<<dim3(HH / 64, M2 / 256), 256, 0, stream>>>(
      gf2, dWT, nullptr, nullptr, t2, nullptr, M2, HH, HH);
  gemm_wmma_kernel<1><<<dim3(HH / 64, M2 / 256), 256, 0, stream>>>(
      t2, g1T, g1_b, nullptr, u2, nullptr, M2, HH, HH);
  gemm_wmma_kernel<0><<<dim3(HH / 64, M2 / 256), 256, 0, stream>>>(
      u2, g2T, g2_b, nullptr, h2, nullptr, M2, HH, HH);

  // ---- v = x @ wv ----
  gemm_wmma_kernel<0><<<dim3(HH / 64, NN / 256), 256, 0, stream>>>(
      x16, wvT, nullptr, nullptr, v16, nullptr, NN, HH, HH);

  // ---- column softmax stats (axis=0) ----
  colstats_kernel<<<HH, 256, 0, stream>>>(h1, NN, HH, 0.08838834764831845f, cmax1, csum1);
  colstats_kernel<<<HH, 256, 0, stream>>>(h2, M2, HH, 0.03125f, cmax2, csum2);

  // ---- fused attn + PE + res ----
  fuse_res_kernel<<<(int)(((size_t)NN * HH + 255) / 256), 256, 0, stream>>>(
      h1, h2, v16, cmax1, csum1, cmax2, csum2, features, lpe, res16, NN, HH, CC);

  // ---- out = res @ fc2_w + fc2_b + features ----
  gemm_wmma_kernel<2><<<dim3(CC / 64, NN / 256), 256, 0, stream>>>(
      res16, fc2wT, fc2_b, features, nullptr, out, NN, HH, CC);
}
